// TimeLSTM_44959717654702
// MI455X (gfx1250) — compile-verified, matrix operand order checked by reading, not run on hardware
//
#include <hip/hip_runtime.h>
#include <math.h>

// ---------------------------------------------------------------------------
// TimeLSTM for MI455X (gfx1250): bf16 WMMA GEMMs + persistent-LDS scan kernel.
// B=64, S=512, E_MOD=768, E=H=512, 4H=2048.
// ---------------------------------------------------------------------------

typedef __attribute__((ext_vector_type(16))) __bf16 v16bf;
typedef __attribute__((ext_vector_type(8)))  float  v8f;

#define BB   64
#define SS   512
#define EM   768
#define HH   512
#define FH   2048

__device__ __forceinline__ v8f wmma_bf16(v16bf a, v16bf b, v8f c) {
    return __builtin_amdgcn_wmma_f32_16x16x32_bf16(false, a, false, b, (short)0, c, false, false);
}

__device__ __forceinline__ float sigmoidf_(float x) { return 1.0f / (1.0f + __expf(-x)); }

// Load one 16x32 bf16 A-fragment from a row-major fp32 matrix (lda in floats),
// converting in registers. Layout per CDNA5 ISA ("16-bit A 16x32"):
//   lane = m + 16*half,  idx = sub + 8*group,  k = group*16 + half*8 + sub
// => each lane reads two contiguous runs of 8 floats.
__device__ __forceinline__ v16bf load_a_frag(const float* __restrict__ A, int lda,
                                             int row0, int k0, int lane) {
    const int m    = row0 + (lane & 15);
    const int half = lane >> 4;
    const float* p = A + (size_t)m * lda + k0 + half * 8;
    const float4 f0 = *(const float4*)(p);
    const float4 f1 = *(const float4*)(p + 4);
    const float4 f2 = *(const float4*)(p + 16);
    const float4 f3 = *(const float4*)(p + 20);
    v16bf a;
    a[0]=(__bf16)f0.x;  a[1]=(__bf16)f0.y;  a[2]=(__bf16)f0.z;  a[3]=(__bf16)f0.w;
    a[4]=(__bf16)f1.x;  a[5]=(__bf16)f1.y;  a[6]=(__bf16)f1.z;  a[7]=(__bf16)f1.w;
    a[8]=(__bf16)f2.x;  a[9]=(__bf16)f2.y;  a[10]=(__bf16)f2.z; a[11]=(__bf16)f2.w;
    a[12]=(__bf16)f3.x; a[13]=(__bf16)f3.y; a[14]=(__bf16)f3.z; a[15]=(__bf16)f3.w;
    return a;
}

// ---------------------------------------------------------------------------
// Convert fp32 weight W[K x N] (row-major) into bf16 B-fragments, pre-swizzled
// so that lane L of a wave reads its 16 bf16 values contiguously at
// frag_base + L*16.  frag index = nt*KT + kt (1 KB per fragment).
// ---------------------------------------------------------------------------
__global__ void k_convert_swz(const float* __restrict__ W, __bf16* __restrict__ dst,
                              int K, int N) {
    int e = blockIdx.x * blockDim.x + threadIdx.x;
    if (e >= K * N) return;
    int k = e / N, n = e - k * N;
    int nt = n >> 4, n_in = n & 15;
    int kt = k >> 5, k_in = k & 31;
    int group = k_in >> 4, within = k_in & 15;
    int half = within >> 3, sub = within & 7;
    int lane = n_in + (half << 4);
    int idx  = sub + (group << 3);
    int KT   = K >> 5;
    dst[(((size_t)nt * KT + kt) << 9) + lane * 16 + idx] = (__bf16)W[e];
}

// Zero the h/c double buffers (4 slabs of B*H fp32) and the grid barrier.
__global__ void k_zero_init(float* __restrict__ hc, unsigned* __restrict__ bar) {
    int i = blockIdx.x * blockDim.x + threadIdx.x;     // grid sized to 4*B*H
    hc[i] = 0.0f;
    if (i == 0) *bar = 0u;
}

// ---------------------------------------------------------------------------
// C[M x N] = A[M x K] (fp32, row-major) * Bsw (bf16 swizzled) + bias, fp32 out.
// 64x64 tile per block, 256 threads (8 wave32), 2 accumulators per wave.
// ---------------------------------------------------------------------------
__global__ void k_wmma_gemm_bias(const float* __restrict__ A,
                                 const __bf16* __restrict__ Bsw,
                                 const float* __restrict__ bias,
                                 float* __restrict__ C, int N, int K) {
    const int KT   = K >> 5;
    const int tid  = threadIdx.x;
    const int wave = tid >> 5, lane = tid & 31;
    const int mt   = wave >> 1;                 // 0..3
    const int ntB  = (wave & 1) * 2;            // 0 or 2
    const int row0 = blockIdx.x * 64 + mt * 16;
    const int ntg0 = blockIdx.y * 4 + ntB;

    v8f acc0 = {}, acc1 = {};
    for (int kt = 0; kt < KT; ++kt) {
        v16bf a  = load_a_frag(A, K, row0, kt << 5, lane);
        v16bf b0 = *(const v16bf*)(Bsw + (((size_t)ntg0       * KT + kt) << 9) + lane * 16);
        v16bf b1 = *(const v16bf*)(Bsw + (((size_t)(ntg0 + 1) * KT + kt) << 9) + lane * 16);
        acc0 = wmma_bf16(a, b0, acc0);
        acc1 = wmma_bf16(a, b1, acc1);
    }
    const int m0 = row0 + ((lane >> 4) << 3);
    const int n0 = blockIdx.y * 64 + ntB * 16 + (lane & 15);
    const float bia0 = bias[n0], bia1 = bias[n0 + 16];
#pragma unroll
    for (int i = 0; i < 8; ++i) {
        C[(size_t)(m0 + i) * N + n0]      = acc0[i] + bia0;
        C[(size_t)(m0 + i) * N + n0 + 16] = acc1[i] + bia1;
    }
}

// ---------------------------------------------------------------------------
// Persistent scan kernel: 16 blocks x 256 threads. Block `blk` owns 32 columns
// (2 N-fragments) of each of the 5 per-step GEMMs (c@W_d and the 4 gates of
// h@W_all). Weight slices (160 KB bf16) are staged once into LDS and stay
// resident for all 512 steps. Each step, h (fp32, global) is cooperatively
// converted once into bf16 A-fragments in LDS (64 KB) and reused by all 32
// gate tiles (8x fewer global fetches + conversions on the serial path).
// h/c double-buffered in global (L2-resident); device-wide sync via monotone
// atomic counter + s_sleep spin.
// Dynamic LDS: 160 KB weights + 64 KB h-fragments + 40 KB scratch = 270336 B.
// ---------------------------------------------------------------------------
__global__ void k_scan(float* __restrict__ hc,          // [4][B*H]: h0,h1,c0,c1
                       const float* __restrict__ u,     // [B*S, 4H]
                       const float* __restrict__ tim,   // [B, S]
                       const float* __restrict__ b_all, // [4H]
                       const float* __restrict__ b_d,   // [H]
                       const __bf16* __restrict__ wall_sw, // swizzled [512 x 2048]
                       const __bf16* __restrict__ wd_sw,   // swizzled [512 x 512]
                       unsigned* __restrict__ bar) {
    extern __shared__ char smem[];
    __bf16* lds_w = (__bf16*)smem;                       // 160 frags * 1 KB
    __bf16* lds_h = (__bf16*)(smem + 160 * 1024);        // 64 frags * 1 KB
    float*  scr   = (float*)(smem + 160 * 1024 + 64 * 1024); // [5][64][32]

    const int blk = blockIdx.x;        // 0..15
    const int tid = threadIdx.x;
    const int wave = tid >> 5, lane = tid & 31;

    // --- stage resident weight fragments: mat 0 = W_d slice, 1..4 = gates ---
    for (int f = 0; f < 160; ++f) {
        int mat = f >> 5, rem = f & 31, l = rem >> 4, kt = rem & 15;
        const __bf16* src;
        if (mat == 0) {
            src = wd_sw + (((size_t)(blk * 2 + l) * 16 + kt) << 9);
        } else {
            int ntg = (mat - 1) * 32 + blk * 2 + l;     // gate*32 + local
            src = wall_sw + (((size_t)ntg * 16 + kt) << 9);
        }
        ((unsigned*)(lds_w))[f * 256 + tid] = ((const unsigned*)src)[tid];
    }
    __syncthreads();

    for (int s = 0; s < SS; ++s) {
        const float* h_cur = hc + (size_t)(s & 1) * (BB * HH);
        float*       h_nxt = hc + (size_t)((s + 1) & 1) * (BB * HH);
        const float* c_cur = hc + (size_t)(2 + (s & 1)) * (BB * HH);
        float*       c_nxt = hc + (size_t)(2 + ((s + 1) & 1)) * (BB * HH);

        // ---- stage h as bf16 A-fragments in LDS (frag = mt*16 + kt) ----
        // 4096 runs of 8 contiguous k-values; run -> (frag, lane_, group).
        for (int r = tid; r < 4096; r += 256) {
            const int frag = r >> 6, rr = r & 63;
            const int ln = rr >> 1, g = rr & 1;
            const int m = ((frag >> 4) << 4) + (ln & 15);
            const int k = ((frag & 15) << 5) + (g << 4) + ((ln >> 4) << 3);
            const float4 fa = *(const float4*)(h_cur + (size_t)m * HH + k);
            const float4 fb = *(const float4*)(h_cur + (size_t)m * HH + k + 4);
            __bf16* d = lds_h + (frag << 9) + ln * 16 + g * 8;
            d[0]=(__bf16)fa.x; d[1]=(__bf16)fa.y; d[2]=(__bf16)fa.z; d[3]=(__bf16)fa.w;
            d[4]=(__bf16)fb.x; d[5]=(__bf16)fb.y; d[6]=(__bf16)fb.z; d[7]=(__bf16)fb.w;
        }
        __syncthreads();

        // ---- 40 output tiles (5 mats x mt(4) x nt(2)), 5 per wave ----
        for (int q = 0; q < 5; ++q) {
            const int t   = wave * 5 + q;
            const int mat = t >> 3, r = t & 7, mt = r >> 1, nt = r & 1;
            v8f acc = {};
#pragma unroll 4
            for (int kt = 0; kt < 16; ++kt) {
                v16bf a;
                if (mat == 0) {  // c @ W_d tile: direct global + convert
                    a = load_a_frag(c_cur, HH, mt * 16, kt << 5, lane);
                } else {         // gate tiles: shared bf16 fragments from LDS
                    a = *(const v16bf*)(lds_h + (((mt << 4) + kt) << 9) + lane * 16);
                }
                v16bf b = *(const v16bf*)(lds_w + ((((mat * 2 + nt) * 16 + kt) << 9) + lane * 16));
                acc = wmma_bf16(a, b, acc);
            }
            const int nloc = nt * 16 + (lane & 15);
#pragma unroll
            for (int i = 0; i < 8; ++i) {
                const int m = mt * 16 + ((lane >> 4) << 3) + i;   // batch index
                float v = acc[i];
                if (mat == 0) {
                    v = tanhf(v + b_d[blk * 32 + nloc]);          // c_s1 tile
                } else {
                    const int col = (mat - 1) * HH + blk * 32 + nloc;
                    v += b_all[col] + u[((size_t)m * SS + s) * FH + col];
                }
                scr[(mat * 64 + m) * 32 + nloc] = v;
            }
        }
        __syncthreads();

        // ---- fused gate math for this block's 64x32 slice ----
        for (int e = tid; e < 64 * 32; e += 256) {
            const int m = e >> 5, n = e & 31, col = blk * 32 + n;
            const float cs1 = scr[(0 * 64 + m) * 32 + n];
            const float fg  = sigmoidf_(scr[(1 * 64 + m) * 32 + n]);
            const float ig  = sigmoidf_(scr[(2 * 64 + m) * 32 + n]);
            const float og  = sigmoidf_(scr[(3 * 64 + m) * 32 + n]);
            const float cg  = sigmoidf_(scr[(4 * 64 + m) * 32 + n]);
            const float g   = 1.0f / (tim[m * SS + s] + 1.0f);
            const float co  = c_cur[(size_t)m * HH + col];
            const float cadj = (co - cs1) + cs1 * g;
            const float cn  = fg * cadj + ig * cg;
            c_nxt[(size_t)m * HH + col] = cn;
            h_nxt[(size_t)m * HH + col] = og * tanhf(cn);
        }
        __threadfence();
        __syncthreads();

        // ---- device-wide barrier (monotone counter, 16 arrivals/step) ----
        if (tid == 0) {
            atomicAdd(bar, 1u);
            const unsigned tgt = 16u * (unsigned)(s + 1);
            while (__hip_atomic_load(bar, __ATOMIC_RELAXED, __HIP_MEMORY_SCOPE_AGENT) < tgt) {
                __builtin_amdgcn_s_sleep(1);
            }
        }
        __syncthreads();
        __threadfence();
    }
}

// ---------------------------------------------------------------------------
// Output head: hid = gelu_exact(h_last @ Wo1 + bo1); logits = hid @ Wo2 + bo2
// ---------------------------------------------------------------------------
__global__ void k_head1(const float* __restrict__ hL, const float* __restrict__ Wo1,
                        const float* __restrict__ bo1, float* __restrict__ hid) {
    __shared__ float hrow[HH];
    const int b = blockIdx.x, tid = threadIdx.x;
    hrow[tid]       = hL[(size_t)b * HH + tid];
    hrow[tid + 256] = hL[(size_t)b * HH + tid + 256];
    __syncthreads();
#pragma unroll
    for (int half = 0; half < 2; ++half) {
        const int j = tid + half * 256;
        float acc = bo1[j];
        for (int k = 0; k < HH; ++k) acc = fmaf(hrow[k], Wo1[(size_t)k * HH + j], acc);
        // exact erf GELU
        hid[(size_t)b * HH + j] = 0.5f * acc * (1.0f + erff(acc * 0.70710678118654752f));
    }
}

__global__ void k_head2(const float* __restrict__ hid, const float* __restrict__ Wo2,
                        const float* __restrict__ bo2, float* __restrict__ out) {
    __shared__ float red[256];
    const int b = blockIdx.x, tid = threadIdx.x;
    float acc = 0.0f;
    for (int k = tid; k < HH; k += 256) acc = fmaf(hid[(size_t)b * HH + k], Wo2[k], acc);
    red[tid] = acc;
    __syncthreads();
    for (int st = 128; st > 0; st >>= 1) {
        if (tid < st) red[tid] += red[tid + st];
        __syncthreads();
    }
    if (tid == 0) {
        const float logit = red[0] + bo2[0];
        out[b]      = logit;                          // logits [B,1]
        out[BB + b] = 1.0f / (1.0f + __expf(-logit)); // sigmoid(logits)
    }
}

// ---------------------------------------------------------------------------
extern "C" void kernel_launch(void* const* d_in, const int* in_sizes, int n_in,
                              void* d_out, int out_size, void* d_ws, size_t ws_size,
                              hipStream_t stream) {
    const float* modality = (const float*)d_in[0];
    const float* tim      = (const float*)d_in[1];
    const float* Wp       = (const float*)d_in[2];
    const float* bp       = (const float*)d_in[3];
    const float* W_all    = (const float*)d_in[4];
    const float* b_all    = (const float*)d_in[5];
    const float* U_all    = (const float*)d_in[6];
    const float* b_u      = (const float*)d_in[7];
    const float* W_d      = (const float*)d_in[8];
    const float* b_d      = (const float*)d_in[9];
    const float* Wo1      = (const float*)d_in[10];
    const float* bo1      = (const float*)d_in[11];
    const float* Wo2      = (const float*)d_in[12];
    const float* bo2      = (const float*)d_in[13];
    float* out = (float*)d_out;

    char* ws = (char*)d_ws;
    size_t off = 0;
    auto take = [&](size_t bytes) { size_t o = off; off += (bytes + 255) & ~(size_t)255; return o; };

    __bf16* wp_sw   = (__bf16*)(ws + take((size_t)EM * HH * 2));
    __bf16* uall_sw = (__bf16*)(ws + take((size_t)HH * FH * 2));
    __bf16* wall_sw = (__bf16*)(ws + take((size_t)HH * FH * 2));
    __bf16* wd_sw   = (__bf16*)(ws + take((size_t)HH * HH * 2));
    float*  xbuf    = (float*) (ws + take((size_t)BB * SS * HH * 4));
    float*  ubuf    = (float*) (ws + take((size_t)BB * SS * FH * 4));
    float*  hc      = (float*) (ws + take((size_t)4 * BB * HH * 4));
    float*  hid     = (float*) (ws + take((size_t)BB * HH * 4));
    unsigned* bar   = (unsigned*)(ws + take(256));
    (void)ws_size; (void)in_sizes; (void)n_in; (void)out_size;

    // weight conversion + swizzle to WMMA B-fragment layout
    k_convert_swz<<<dim3((EM * HH) / 256), 256, 0, stream>>>(Wp,    wp_sw,   EM, HH);
    k_convert_swz<<<dim3((HH * FH) / 256), 256, 0, stream>>>(U_all, uall_sw, HH, FH);
    k_convert_swz<<<dim3((HH * FH) / 256), 256, 0, stream>>>(W_all, wall_sw, HH, FH);
    k_convert_swz<<<dim3((HH * HH) / 256), 256, 0, stream>>>(W_d,   wd_sw,   HH, HH);
    k_zero_init<<<dim3((4 * BB * HH) / 256), 256, 0, stream>>>(hc, bar);

    // x = modality @ Wp + bp   ([32768 x 768] x [768 x 512])
    k_wmma_gemm_bias<<<dim3((BB * SS) / 64, HH / 64), 256, 0, stream>>>(modality, wp_sw, bp, xbuf, HH, EM);
    // u = x @ U_all + b_u      ([32768 x 512] x [512 x 2048])
    k_wmma_gemm_bias<<<dim3((BB * SS) / 64, FH / 64), 256, 0, stream>>>(xbuf, uall_sw, b_u, ubuf, FH, HH);

    // persistent recurrent scan (16 co-resident WGPs, 264 KB LDS each)
    static const int kScanLds = 160 * 1024 + 64 * 1024 + 5 * 64 * 32 * 4; // 270336 B
    hipFuncSetAttribute((const void*)k_scan, hipFuncAttributeMaxDynamicSharedMemorySize, kScanLds);
    k_scan<<<dim3(16), 256, kScanLds, stream>>>(hc, ubuf, tim, b_all, b_d, wall_sw, wd_sw, bar);

    // output head; final h lands in slab 0 of hc after 512 steps
    k_head1<<<dim3(BB), 256, 0, stream>>>(hc, Wo1, bo1, hid);
    k_head2<<<dim3(BB), 256, 0, stream>>>(hid, Wo2, bo2, out);
}
